// TorchRecDLRM_7413113552923
// MI455X (gfx1250) — matrix operand order repaired and checked by reading, not compiled, not run on hardware
//
#include <hip/hip_runtime.h>
#include <hip/hip_bf16.h>

// ---------------------------------------------------------------------------
// DLRM forward for MI455X (gfx1250): bf16 WMMA GEMMs + coalesced gather pooling
// GEMM: one wave computes a 32x64 C macro-tile (2x4 WMMA tiles) so each A
// fragment is reused 4x and each B fragment 2x -> 1.5 b128 loads per WMMA.
// ---------------------------------------------------------------------------

typedef __bf16 bf16_t;
typedef __attribute__((ext_vector_type(16))) __bf16 v16bf;
typedef __attribute__((ext_vector_type(8)))  __bf16 v8bf;
typedef __attribute__((ext_vector_type(8)))  float  v8f;

union V16U { v16bf v; v8bf h[2]; };

#define B_SZ   16384
#define F_SZ   26
#define N_ROWS 100000
#define D_SZ   64
#define L_SZ   4
#define NFEAT  27          // 1 dense + 26 sparse
#define NPAIR  351         // 27*26/2
#define TOPK   416         // 64 + 351 padded to mult of 32

__device__ __forceinline__ v8f zero8f() {
    v8f z = {0.f,0.f,0.f,0.f,0.f,0.f,0.f,0.f};
    return z;
}
__device__ __forceinline__ v8bf zero8bf() {
    v8bf z;
#pragma unroll
    for (int i = 0; i < 8; ++i) z[i] = (__bf16)0.f;
    return z;
}

// --------------------------- weight convert/transpose -----------------------
// W is K x N row-major fp32; Wt is N x Kpad row-major bf16, zero-padded K.
__global__ __launch_bounds__(256) void wt_convert(const float* __restrict__ W,
                                                  bf16_t* __restrict__ Wt,
                                                  int K, int N, int Kpad) {
    int idx = blockIdx.x * 256 + threadIdx.x;
    if (idx >= N * Kpad) return;
    int n = idx / Kpad, kp = idx - n * Kpad;
    float v = (kp < K) ? W[(size_t)kp * N + n] : 0.f;
    Wt[idx] = (bf16_t)v;
}

// dense (B x 13 fp32) -> bf16 padded to B x 32
__global__ __launch_bounds__(256) void dense_pad(const float* __restrict__ d,
                                                 bf16_t* __restrict__ dp) {
    int idx = blockIdx.x * 256 + threadIdx.x;      // B*32 threads
    int b = idx >> 5, k = idx & 31;
    dp[idx] = (k < 13) ? (bf16_t)d[b * 13 + k] : (bf16_t)0.f;
}

// ------------------------------- WMMA GEMM ----------------------------------
// C[M x N] = relu?(A[M x K](bf16, lda) @ Wt^T + bias), Wt is N x K bf16.
// grid = (N/64, M/128), block = 128 (4 waves).
// One wave -> 32(M) x 64(N) C macro-tile = 2x4 WMMA tiles; A frags reused 4x,
// B frags reused 2x.
__global__ __launch_bounds__(128) void gemm_bf16_wmma(
        const bf16_t* __restrict__ A, int lda,
        const bf16_t* __restrict__ Wt,
        const float*  __restrict__ bias,
        bf16_t* __restrict__ C, int ldc,
        int K, int relu) {
    const int lane = threadIdx.x & 31;
    const int wave = threadIdx.x >> 5;
    const int tn = blockIdx.x;                 // 64-wide N macro-tile
    const int tm = blockIdx.y * 4 + wave;      // 32-high M macro-tile
    const int l15 = lane & 15;
    const int hi  = lane >> 4;                 // 0 or 1
    const int aoff = hi ? 16 : 0;              // A: lanes16-31 hold K 16..31
    const int boff = hi ? 16 : 0;              // B: lanes16-31 hold K 16..31

    const bf16_t* arow0 = A + (size_t)(tm * 32 + l15)      * lda;   // M rows
    const bf16_t* arow1 = A + (size_t)(tm * 32 + 16 + l15) * lda;
    const bf16_t* brow[4];
#pragma unroll
    for (int j = 0; j < 4; ++j)
        brow[j] = Wt + (size_t)(tn * 64 + j * 16 + l15) * K;        // N cols

    v8f acc[2][4];
#pragma unroll
    for (int i = 0; i < 2; ++i)
#pragma unroll
        for (int j = 0; j < 4; ++j) acc[i][j] = zero8f();

    for (int k = 0; k < K; k += 32) {
        V16U a[2], b[4];
        a[0].h[0] = *(const v8bf*)(arow0 + k + aoff);
        a[0].h[1] = *(const v8bf*)(arow0 + k + aoff + 8);
        a[1].h[0] = *(const v8bf*)(arow1 + k + aoff);
        a[1].h[1] = *(const v8bf*)(arow1 + k + aoff + 8);
#pragma unroll
        for (int j = 0; j < 4; ++j)
            b[j].v = *(const v16bf*)(brow[j] + k + boff);
#pragma unroll
        for (int i = 0; i < 2; ++i)
#pragma unroll
            for (int j = 0; j < 4; ++j)
                acc[i][j] = __builtin_amdgcn_wmma_f32_16x16x32_bf16(
                    false, a[i].v, false, b[j].v, (short)0, acc[i][j],
                    false, false);
    }

#pragma unroll
    for (int j = 0; j < 4; ++j) {
        const int col = tn * 64 + j * 16 + l15;
        const float bv = bias[col];
#pragma unroll
        for (int i = 0; i < 2; ++i) {
#pragma unroll
            for (int r = 0; r < 8; ++r) {
                int m = tm * 32 + i * 16 + r + hi * 8;  // C layout: VGPR r -> row
                float v = acc[i][j][r] + bv;
                if (relu && v < 0.f) v = 0.f;
                C[(size_t)m * ldc + col] = (bf16_t)v;
            }
        }
    }
}

// --------------------------- embedding pooling ------------------------------
// grid = (B/8, F), block = 256 (8 waves), one wave -> one (f, b) pooled row.
__global__ __launch_bounds__(256) void embed_pool(
        const long long* __restrict__ idx,     // F x B x L (int64)
        const float* __restrict__ tables,      // F x N x D
        bf16_t* __restrict__ feats) {          // B x 27 x 64
    const int lane = threadIdx.x & 31;
    const int wave = threadIdx.x >> 5;
    const int b = blockIdx.x * 8 + wave;
    const int f = blockIdx.y;
    const long long* ip = idx + ((size_t)f * B_SZ + b) * L_SZ;
    const float* tab = tables + (size_t)f * N_ROWS * D_SZ;
    const int e = lane * 2;
    float s0 = 0.f, s1 = 0.f;
#pragma unroll
    for (int l = 0; l < L_SZ; ++l) {
        const float* r = tab + (size_t)ip[l] * D_SZ + e;
        s0 += r[0];
        s1 += r[1];
    }
    bf16_t* o = feats + ((size_t)b * NFEAT + 1 + f) * D_SZ + e;
    o[0] = (bf16_t)s0;
    o[1] = (bf16_t)s1;
}

// ------------------------------ interaction ---------------------------------
// One wave per sample: Z = feats @ feats^T (27x27, padded to 32x32 via zeroed
// fragments), 8 x v_wmma_f32_16x16x32_bf16, scatter upper-tri pairs to y0.
// Also copies dense slot -> y0[0:64] and zeroes pad col 415.
// grid = B/4, block = 128.
__global__ __launch_bounds__(128) void interact(
        const bf16_t* __restrict__ feats,      // B x 27 x 64
        bf16_t* __restrict__ y0) {             // B x 416
    const int lane = threadIdx.x & 31;
    const int wave = threadIdx.x >> 5;
    const int b = blockIdx.x * 4 + wave;
    const bf16_t* fb = feats + (size_t)b * NFEAT * D_SZ;
    bf16_t* yb = y0 + (size_t)b * TOPK;
    const int l15 = lane & 15;
    const int hi  = lane >> 4;

    // dense copy + pad
    {
        int e = lane * 2;
        yb[e]     = fb[e];
        yb[e + 1] = fb[e + 1];
        if (lane == 0) yb[415] = (bf16_t)0.f;
    }

    v8f acc[2][2];
    acc[0][0] = zero8f(); acc[0][1] = zero8f();
    acc[1][0] = zero8f(); acc[1][1] = zero8f();

    for (int kk = 0; kk < D_SZ; kk += 32) {
        V16U a[2], bm[2];
#pragma unroll
        for (int t = 0; t < 2; ++t) {
            const int rowA = t * 16 + l15;     // A-matrix row
            if (rowA < NFEAT) {
                const bf16_t* ar = fb + rowA * D_SZ + kk;
                const int aoff = hi ? 16 : 0;
                a[t].h[0] = *(const v8bf*)(ar + aoff);
                a[t].h[1] = *(const v8bf*)(ar + aoff + 8);
            } else {
                a[t].h[0] = zero8bf();
                a[t].h[1] = zero8bf();
            }
            const int colB = t * 16 + l15;     // B-matrix column = feats row
            if (colB < NFEAT) {
                const bf16_t* br = fb + colB * D_SZ + kk + (hi ? 16 : 0);
                bm[t].v = *(const v16bf*)br;
            } else {
                bm[t].h[0] = zero8bf();
                bm[t].h[1] = zero8bf();
            }
        }
#pragma unroll
        for (int ti = 0; ti < 2; ++ti)
#pragma unroll
            for (int tj = 0; tj < 2; ++tj)
                acc[ti][tj] = __builtin_amdgcn_wmma_f32_16x16x32_bf16(
                    false, a[ti].v, false, bm[tj].v, (short)0, acc[ti][tj],
                    false, false);
    }

    // scatter upper-triangle pairs (row-major (i,j), j>i) straight from C layout
#pragma unroll
    for (int ti = 0; ti < 2; ++ti)
#pragma unroll
        for (int tj = 0; tj < 2; ++tj)
#pragma unroll
            for (int r = 0; r < 8; ++r) {
                int i = ti * 16 + r + hi * 8;
                int j = tj * 16 + l15;
                if (j > i && j < NFEAT) {
                    int p = i * (F_SZ) - (i * (i - 1)) / 2 + (j - i - 1);
                    yb[64 + p] = (bf16_t)acc[ti][tj][r];
                }
            }
}

// ----------------------------- final 256 -> 1 -------------------------------
__global__ __launch_bounds__(256) void top_final(const bf16_t* __restrict__ t2,
                                                 const bf16_t* __restrict__ w2,
                                                 const float* __restrict__ b2,
                                                 float* __restrict__ out) {
    int i = blockIdx.x * 256 + threadIdx.x;
    if (i >= B_SZ) return;
    const bf16_t* r = t2 + (size_t)i * 256;
    float s = b2[0];
#pragma unroll 8
    for (int k = 0; k < 256; ++k) s += (float)r[k] * (float)w2[k];
    out[i] = s;
}

// ---------------------------------------------------------------------------
extern "C" void kernel_launch(void* const* d_in, const int* in_sizes, int n_in,
                              void* d_out, int out_size, void* d_ws, size_t ws_size,
                              hipStream_t stream) {
    const float*     dense  = (const float*)d_in[0];
    const long long* sp_idx = (const long long*)d_in[1];   // int64 indices
    const float*     tables = (const float*)d_in[2];
    const float* bot_W0 = (const float*)d_in[3];
    const float* bot_b0 = (const float*)d_in[4];
    const float* bot_W1 = (const float*)d_in[5];
    const float* bot_b1 = (const float*)d_in[6];
    const float* bot_W2 = (const float*)d_in[7];
    const float* bot_b2 = (const float*)d_in[8];
    const float* top_W0 = (const float*)d_in[9];
    const float* top_b0 = (const float*)d_in[10];
    const float* top_W1 = (const float*)d_in[11];
    const float* top_b1 = (const float*)d_in[12];
    const float* top_W2 = (const float*)d_in[13];
    const float* top_b2 = (const float*)d_in[14];
    float* out = (float*)d_out;

    // ---- workspace carve-up (256B aligned) ----
    char* ws = (char*)d_ws;
    size_t off = 0;
    auto alloc = [&](size_t bytes) -> char* {
        char* p = ws + off;
        off += (bytes + 255) & ~(size_t)255;
        return p;
    };
    bf16_t* dp    = (bf16_t*)alloc((size_t)B_SZ * 32 * 2);          // padded dense
    bf16_t* h0    = (bf16_t*)alloc((size_t)B_SZ * 512 * 2);         // also t1
    bf16_t* h1    = (bf16_t*)alloc((size_t)B_SZ * 256 * 2);         // also t2
    bf16_t* feats = (bf16_t*)alloc((size_t)B_SZ * NFEAT * D_SZ * 2);
    bf16_t* y0    = (bf16_t*)alloc((size_t)B_SZ * TOPK * 2);
    bf16_t* wt_b0 = (bf16_t*)alloc((size_t)512 * 32 * 2);
    bf16_t* wt_b1 = (bf16_t*)alloc((size_t)256 * 512 * 2);
    bf16_t* wt_b2 = (bf16_t*)alloc((size_t)64 * 256 * 2);
    bf16_t* wt_t0 = (bf16_t*)alloc((size_t)512 * TOPK * 2);
    bf16_t* wt_t1 = (bf16_t*)alloc((size_t)256 * 512 * 2);
    bf16_t* w2    = (bf16_t*)alloc((size_t)256 * 2);
    (void)ws_size; (void)in_sizes; (void)n_in; (void)out_size;

    // ---- weight conversion (tiny) ----
    auto cvt = [&](const float* W, bf16_t* Wt, int K, int N, int Kpad) {
        int n = N * Kpad;
        wt_convert<<<(n + 255) / 256, 256, 0, stream>>>(W, Wt, K, N, Kpad);
    };
    cvt(bot_W0, wt_b0, 13, 512, 32);
    cvt(bot_W1, wt_b1, 512, 256, 512);
    cvt(bot_W2, wt_b2, 256, 64, 256);
    cvt(top_W0, wt_t0, 415, 512, TOPK);
    cvt(top_W1, wt_t1, 512, 256, 512);
    cvt(top_W2, w2, 256, 1, 256);

    dense_pad<<<(B_SZ * 32) / 256, 256, 0, stream>>>(dense, dp);

    // ---- bottom MLP ----
    gemm_bf16_wmma<<<dim3(512 / 64, B_SZ / 128), 128, 0, stream>>>(
        dp, 32, wt_b0, bot_b0, h0, 512, 32, 1);
    gemm_bf16_wmma<<<dim3(256 / 64, B_SZ / 128), 128, 0, stream>>>(
        h0, 512, wt_b1, bot_b1, h1, 256, 512, 1);
    // bottom output goes directly into feats slot 0 (ldc = 27*64)
    gemm_bf16_wmma<<<dim3(64 / 64, B_SZ / 128), 128, 0, stream>>>(
        h1, 256, wt_b2, bot_b2, feats, NFEAT * D_SZ, 256, 1);

    // ---- embedding pooling into feats slots 1..26 ----
    embed_pool<<<dim3(B_SZ / 8, F_SZ), 256, 0, stream>>>(sp_idx, tables, feats);

    // ---- pairwise interaction -> y0 (dense copy + 351 pairs + pad) ----
    interact<<<B_SZ / 4, 128, 0, stream>>>(feats, y0);

    // ---- top MLP (reuse h0/h1 as t1/t2) ----
    gemm_bf16_wmma<<<dim3(512 / 64, B_SZ / 128), 128, 0, stream>>>(
        y0, TOPK, wt_t0, top_b0, h0, 512, TOPK, 1);
    gemm_bf16_wmma<<<dim3(256 / 64, B_SZ / 128), 128, 0, stream>>>(
        h0, 512, wt_t1, top_b1, h1, 256, 512, 1);
    top_final<<<B_SZ / 256, 256, 0, stream>>>(h1, w2, top_b2, out);
}
